// SAModule_37641093382402
// MI455X (gfx1250) — matrix operand order repaired and checked by reading, not compiled, not run on hardware
//
#include <hip/hip_runtime.h>
#include <hip/hip_bf16.h>

// ---------------------------------------------------------------------------
// SAModule (FPS + radius PointNetConv) for MI455X / gfx1250, wave32 + WMMA.
//
//   1) fps_kernel : sequential farthest-point sampling (1 block / batch);
//                   per-wave shuffle argmax (no barriers) + 1 LDS combine.
//   2) precompA   : A[b,j,:] = x_j@W1x + pos_j@W1p + b1  (layer-1 folded:
//                   h1[i,j] = ReLU(A_j - c_i), c_i = q_i@W1p).
//   3) samod_main : one wave per query. Ballot-compacted radius list, padded
//                   to 16 with a repeated valid neighbor (duplicates cannot
//                   change the max); per 16-neighbor tile layers 2+3 run on
//                   v_wmma_f32_16x16x32_f16 (f32 accum), ReLU, running max.
//                   NOTE: no min-waves bound — letting the compiler pin the 24
//                   loop-invariant B-fragments in VGPRs (~300 regs, MSB
//                   addressing) beats forcing <=256 which spilled to scratch.
//
// WMMA fragment layouts (ISA 7.12.2 / 7.12.4, wave32, 16-bit):
//   A 16x32 : lane L -> row m=L&15 ; elem e (v=e/2,h=e&1):
//             K = (v<4?0:16) + (L>=16?8:0) + 2*(v&3) + h
//   B 32x16 : lane L -> col n=L&15 ; K = (v<4?0:16) + (L>=16?8:0) + 2*(v&3) + h (as A)
//   C/D 16x16 f32: lane L -> col n=L&15 ; VGPR r -> row m = r + (L>=16?8:0)
// ---------------------------------------------------------------------------

typedef __attribute__((ext_vector_type(16))) _Float16 v16h;
typedef __attribute__((ext_vector_type(8)))  float    v8f;

static constexpr int BB = 2;      // batch
static constexpr int NN = 2048;   // points
static constexpr int FF = 64;     // features
static constexpr int KK = 512;    // sampled queries (N * 0.25)
static constexpr float R2 = 0.04f;
static constexpr int WAVES = 4;   // queries per block

// ---------------------------------------------------------------------------
// 1) Farthest point sampling. Emit cur, then argmax of running min-distance;
//    first-index tie-break like jnp.argmax (lexicographic max-v / min-idx is
//    associative, so shuffle-reduction order is irrelevant).
// ---------------------------------------------------------------------------
__global__ __launch_bounds__(256) void fps_kernel(const float* __restrict__ pos,
                                                  int* __restrict__ selOut,
                                                  float* __restrict__ qposOut) {
  const int b = blockIdx.x;
  const int tid = threadIdx.x;                 // 256 threads, 8 pts each
  const int wave = tid >> 5;
  const int lane = tid & 31;
  __shared__ float swv[8];
  __shared__ int   swi[8];
  __shared__ float curp[3];
  __shared__ int   curS;

  const float* P = pos + (size_t)b * NN * 3;
  float dmin[8];
#pragma unroll
  for (int s = 0; s < 8; ++s) dmin[s] = 3.4e38f;

  int cur = 0;
  for (int k = 0; k < KK; ++k) {
    if (tid == 0) {
      selOut[b * KK + k] = cur;
      float px = P[cur * 3 + 0], py = P[cur * 3 + 1], pz = P[cur * 3 + 2];
      curp[0] = px; curp[1] = py; curp[2] = pz;
      float* qp = qposOut + ((size_t)b * KK + k) * 3;
      qp[0] = px; qp[1] = py; qp[2] = pz;
    }
    __syncthreads();
    const float cx = curp[0], cy = curp[1], cz = curp[2];
    float bv = -1.0f; int bj = 0x7fffffff;
#pragma unroll
    for (int s = 0; s < 8; ++s) {
      const int j = tid + s * 256;
      const float dx = P[j * 3 + 0] - cx;
      const float dy = P[j * 3 + 1] - cy;
      const float dz = P[j * 3 + 2] - cz;
      const float d = dx * dx + dy * dy + dz * dz;
      dmin[s] = fminf(dmin[s], d);
      if (dmin[s] > bv) { bv = dmin[s]; bj = j; }   // j ascending -> first max
    }
    // wave-level argmax, no barriers
#pragma unroll
    for (int off = 16; off; off >>= 1) {
      const float ov = __shfl_xor(bv, off, 32);
      const int   oi = __shfl_xor(bj, off, 32);
      if (ov > bv || (ov == bv && oi < bj)) { bv = ov; bj = oi; }
    }
    if (lane == 0) { swv[wave] = bv; swi[wave] = bj; }
    __syncthreads();
    if (wave == 0) {
      float v = (lane < 8) ? swv[lane] : -1.0f;
      int   ix = (lane < 8) ? swi[lane] : 0x7fffffff;
#pragma unroll
      for (int off = 4; off; off >>= 1) {
        const float ov = __shfl_xor(v, off, 32);
        const int   oi = __shfl_xor(ix, off, 32);
        if (ov > v || (ov == v && oi < ix)) { v = ov; ix = oi; }
      }
      if (lane == 0) curS = ix;
    }
    __syncthreads();
    cur = curS;                                  // next curS write is 2 barriers away
  }
}

// ---------------------------------------------------------------------------
// 2) A[b,j,c] = b1[c] + x[b,j,:]@W1[0:64,c] + pos[b,j,:]@W1[64:67,c]
// ---------------------------------------------------------------------------
__global__ __launch_bounds__(256) void precompA(const float* __restrict__ x,
                                                const float* __restrict__ pos,
                                                const float* __restrict__ W1,
                                                const float* __restrict__ b1,
                                                float* __restrict__ A) {
  const int g = blockIdx.x * blockDim.x + threadIdx.x;
  if (g >= BB * NN * 64) return;
  const int c = g & 63;
  const int p = g >> 6;                         // p = b*N + j
  const float* xr = x + (size_t)p * FF;
  float acc = b1[c];
#pragma unroll 8
  for (int f = 0; f < FF; ++f) acc += xr[f] * W1[f * 64 + c];
  const float* pr = pos + (size_t)p * 3;
  acc += pr[0] * W1[64 * 64 + c] + pr[1] * W1[65 * 64 + c] + pr[2] * W1[66 * 64 + c];
  A[(size_t)p * 64 + c] = acc;
}

// ---------------------------------------------------------------------------
// 3) Main: one wave per query.
// ---------------------------------------------------------------------------
__global__ __launch_bounds__(128) void samod_main(const float* __restrict__ pos,
                                                  const float* __restrict__ W1,
                                                  const float* __restrict__ W2,
                                                  const float* __restrict__ b2,
                                                  const float* __restrict__ W3,
                                                  const float* __restrict__ b3,
                                                  const float* __restrict__ A,
                                                  const int* __restrict__ sel,
                                                  float* __restrict__ out) {
  __shared__ alignas(32) _Float16 w2f[8][32][16];     // W2 B-fragments (8 KB)
  __shared__ alignas(32) _Float16 w3f[16][32][16];    // W3 B-fragments (16 KB)
  __shared__ float b2s[64];
  __shared__ float b3s[128];
  __shared__ alignas(16) float cq[WAVES][64];         // c_i = q_i @ W1p
  __shared__ alignas(16) float h2s[WAVES][16][64];    // layer-2 transpose scratch
  __shared__ unsigned short nlist[WAVES][NN + 16];    // compacted neighbor ids

  const int tid = threadIdx.x;
  const int wave = tid >> 5;
  const int lane = tid & 31;
  const int q = blockIdx.x * WAVES + wave;            // 0 .. B*K-1
  const int b = q >> 9;                               // / K (K=512)
  const int i = q & (KK - 1);

  // ---- block setup: biases + pre-swizzled weight fragments in LDS ----
  for (int s = tid; s < 64; s += 128) b2s[s] = b2[s];
  for (int s = tid; s < 128; s += 128) b3s[s] = b3[s];
  for (int s = tid; s < 8 * 32; s += 128) {           // W2: frag f = t*2 + kchunk
    const int f = s >> 5, L = s & 31;
    const int t = f >> 1, c = f & 1;
    const int n = (t << 4) + (L & 15);
#pragma unroll
    for (int e = 0; e < 16; ++e) {
      const int v = e >> 1, hb = e & 1;
      const int k = (c << 5) + ((v < 4) ? 0 : 16) + ((L >= 16) ? 8 : 0) + ((v & 3) << 1) + hb; // B-layout
      w2f[f][L][e] = (_Float16)W2[k * 64 + n];
    }
  }
  for (int s = tid; s < 16 * 32; s += 128) {          // W3
    const int f = s >> 5, L = s & 31;
    const int t = f >> 1, c = f & 1;
    const int n = (t << 4) + (L & 15);
#pragma unroll
    for (int e = 0; e < 16; ++e) {
      const int v = e >> 1, hb = e & 1;
      const int k = (c << 5) + ((v < 4) ? 0 : 16) + ((L >= 16) ? 8 : 0) + ((v & 3) << 1) + hb;
      w3f[f][L][e] = (_Float16)W3[k * 128 + n];
    }
  }

  // ---- per-wave: query position + c_i vector (before the block barrier) ----
  const int jsel = sel[(b << 9) + i];
  const float qx = pos[((size_t)(b * NN + jsel)) * 3 + 0];
  const float qy = pos[((size_t)(b * NN + jsel)) * 3 + 1];
  const float qz = pos[((size_t)(b * NN + jsel)) * 3 + 2];
  for (int kk = lane; kk < 64; kk += 32)
    cq[wave][kk] = qx * W1[64 * 64 + kk] + qy * W1[65 * 64 + kk] + qz * W1[66 * 64 + kk];
  __syncthreads();

  // ---- neighbor search + stream compaction (ballot/popc, wave32) ----
  const float* Pb = pos + (size_t)b * NN * 3;
  int M = 0;
  for (int base = 0; base < NN; base += 32) {
    const int j = base + lane;
    const float dx = Pb[j * 3 + 0] - qx;
    const float dy = Pb[j * 3 + 1] - qy;
    const float dz = Pb[j * 3 + 2] - qz;
    const bool pred = (dx * dx + dy * dy + dz * dz) < R2;
    const unsigned mask = (unsigned)__ballot(pred);
    if (pred) nlist[wave][M + __popc(mask & ((1u << lane) - 1u))] = (unsigned short)j;
    M += __popc(mask);
  }
  const int Mp = (M + 15) & ~15;                      // M >= 1 (self in radius)
  if (lane < Mp - M) nlist[wave][M + lane] = nlist[wave][0];  // duplicate pad

  float runmax[8];
#pragma unroll
  for (int t = 0; t < 8; ++t) runmax[t] = 0.0f;       // ReLU>=0 & M>=1 -> safe

  const float* Ab = A + (size_t)b * NN * 64;
  const int mh = (lane >= 16) ? 8 : 0;
  const int nc = lane & 15;

  for (int tile = 0; tile < Mp; tile += 16) {
    // layer 1: gather A_j, subtract c_i, ReLU, pack into A-fragments (f32->f16)
    const int jm = nlist[wave][tile + (lane & 15)];
    const float* Ar = Ab + (size_t)jm * 64;
    v16h alo, ahi;
#pragma unroll
    for (int v = 0; v < 8; ++v) {
      const int k0 = ((v < 4) ? 0 : 16) + ((lane >= 16) ? 8 : 0) + ((v & 3) << 1);
      const float2 a0 = *(const float2*)(Ar + k0);
      const float2 c0 = *(const float2*)(&cq[wave][k0]);
      alo[2 * v]     = (_Float16)fmaxf(a0.x - c0.x, 0.0f);
      alo[2 * v + 1] = (_Float16)fmaxf(a0.y - c0.y, 0.0f);
      const float2 a1 = *(const float2*)(Ar + 32 + k0);
      const float2 c1 = *(const float2*)(&cq[wave][32 + k0]);
      ahi[2 * v]     = (_Float16)fmaxf(a1.x - c1.x, 0.0f);
      ahi[2 * v + 1] = (_Float16)fmaxf(a1.y - c1.y, 0.0f);
    }

    // layer 2: [16x64] = h1[16x64] @ W2[64x64]  (4 N-tiles x 2 K-chunks)
#pragma unroll
    for (int t = 0; t < 4; ++t) {
      v8f acc = {};
      const v16h bf0 = *(const v16h*)&w2f[2 * t + 0][lane][0];
      acc = __builtin_amdgcn_wmma_f32_16x16x32_f16(false, alo, false, bf0,
                                                   (short)0, acc, false, false);
      const v16h bf1 = *(const v16h*)&w2f[2 * t + 1][lane][0];
      acc = __builtin_amdgcn_wmma_f32_16x16x32_f16(false, ahi, false, bf1,
                                                   (short)0, acc, false, false);
      const int n = (t << 4) + nc;
      const float bias = b2s[n];
#pragma unroll
      for (int r = 0; r < 8; ++r)
        h2s[wave][r + mh][n] = fmaxf(acc[r] + bias, 0.0f);   // C-layout -> LDS
    }

    // transpose C-layout -> A-layout via per-wave LDS scratch (in-order DS)
    v16h glo, ghi;
    {
      const int m = lane & 15;
#pragma unroll
      for (int v = 0; v < 8; ++v) {
        const int k0 = ((v < 4) ? 0 : 16) + ((lane >= 16) ? 8 : 0) + ((v & 3) << 1);
        const float2 x0 = *(const float2*)&h2s[wave][m][k0];
        glo[2 * v]     = (_Float16)x0.x;
        glo[2 * v + 1] = (_Float16)x0.y;
        const float2 x1 = *(const float2*)&h2s[wave][m][32 + k0];
        ghi[2 * v]     = (_Float16)x1.x;
        ghi[2 * v + 1] = (_Float16)x1.y;
      }
    }

    // layer 3: [16x128] = h2 @ W3, bias+ReLU, fold into running max over rows
#pragma unroll
    for (int t = 0; t < 8; ++t) {
      v8f acc = {};
      const v16h bf0 = *(const v16h*)&w3f[2 * t + 0][lane][0];
      acc = __builtin_amdgcn_wmma_f32_16x16x32_f16(false, glo, false, bf0,
                                                   (short)0, acc, false, false);
      const v16h bf1 = *(const v16h*)&w3f[2 * t + 1][lane][0];
      acc = __builtin_amdgcn_wmma_f32_16x16x32_f16(false, ghi, false, bf1,
                                                   (short)0, acc, false, false);
      const int n = (t << 4) + nc;
      const float bias = b3s[n];
      float mx = runmax[t];
#pragma unroll
      for (int r = 0; r < 8; ++r)
        mx = fmaxf(mx, fmaxf(acc[r] + bias, 0.0f));
      runmax[t] = mx;
    }
  }

  // merge half-wave row groups (lanes L and L^16 hold the same column)
#pragma unroll
  for (int t = 0; t < 8; ++t) {
    const float v = fmaxf(runmax[t], __shfl_xor(runmax[t], 16, 32));
    if (lane < 16) out[(size_t)q * 128 + (t << 4) + lane] = v;
  }
}

// ---------------------------------------------------------------------------
extern "C" void kernel_launch(void* const* d_in, const int* in_sizes, int n_in,
                              void* d_out, int out_size, void* d_ws, size_t ws_size,
                              hipStream_t stream) {
  (void)in_sizes; (void)n_in; (void)out_size; (void)ws_size;
  const float* x   = (const float*)d_in[0];
  const float* pos = (const float*)d_in[1];
  const float* W1  = (const float*)d_in[2];
  const float* b1  = (const float*)d_in[3];
  const float* W2  = (const float*)d_in[4];
  const float* b2  = (const float*)d_in[5];
  const float* W3  = (const float*)d_in[6];
  const float* b3  = (const float*)d_in[7];

  float* out      = (float*)d_out;                       // [B,K,128] then [B,K,3]
  float* qpos_out = out + (size_t)BB * KK * 128;
  float* A        = (float*)d_ws;                        // [B,N,64]
  int*   selIdx   = (int*)((char*)d_ws + (size_t)BB * NN * 64 * sizeof(float));

  fps_kernel<<<BB, 256, 0, stream>>>(pos, selIdx, qpos_out);
  precompA<<<(BB * NN * 64 + 255) / 256, 256, 0, stream>>>(x, pos, W1, b1, A);
  samod_main<<<(BB * KK) / WAVES, 128, 0, stream>>>(pos, W1, W2, b2, W3, b3, A,
                                                    selIdx, out);
}
